// FlowEmbedding_51247549776071
// MI455X (gfx1250) — matrix operand order repaired and checked by reading, not compiled, not run on hardware
//
#include <hip/hip_runtime.h>

typedef unsigned short u16;
typedef unsigned int u32;
typedef __attribute__((ext_vector_type(16))) __bf16 v16bf;
typedef __attribute__((ext_vector_type(8)))  float  v8f;

#define BB 8
#define NN 4096
#define KK 16
#define CC 64
#define CO 128
#define MTOT (BB*NN*KK)   // 524288

__device__ __forceinline__ u16 f2bf(float f) {
    u32 u = __float_as_uint(f);
    u32 r = (u + 0x7FFFu + ((u >> 16) & 1u)) >> 16;
    return (u16)r;
}
__device__ __forceinline__ float bf2f(u32 h) {
    return __uint_as_float(h << 16);
}

union Frag { uint4 q[2]; v16bf v; };

// ---------------------------------------------------------------------------
// K0: pack weights into WMMA B-fragment lane order (bf16), zero stat buffers.
// Channel order for layer 0 input: [feat_g 0..63][f1 64..127][rel_xyz 128..130][0 pad ..159]
// -> original W0 column: i<128 -> 3+i ; 128..130 -> i-128 ; else zero.
// Pack layout: wp[((kt*8+nt)*32+lane)*16 + t] = W[n][kbase+t],
//   n = nt*16 + (lane&15), kbase = kt*32 + (lane>>4)*16.
// ---------------------------------------------------------------------------
__global__ void prep_kernel(const float* __restrict__ W0, const float* __restrict__ W1,
                            const float* __restrict__ W2,
                            u16* __restrict__ wp0, u16* __restrict__ wp1, u16* __restrict__ wp2,
                            float* __restrict__ stats) {
    int tid = threadIdx.x;
    for (int i = tid; i < 3 * 256; i += 256) stats[i] = 0.0f;

    const int TOT0 = 5 * 8 * 32 * 16;
    for (int e = tid; e < TOT0; e += 256) {
        int t = e & 15, L = (e >> 4) & 31, nt = (e >> 9) & 7, kt = e >> 12;
        int i = kt * 32 + (L >> 4) * 16 + t;
        int o = nt * 16 + (L & 15);
        float v = 0.0f;
        if (i < 128)       v = W0[o * 131 + (3 + i)];
        else if (i < 131)  v = W0[o * 131 + (i - 128)];
        wp0[e] = f2bf(v);
    }
    const int TOT1 = 4 * 8 * 32 * 16;
    for (int e = tid; e < TOT1; e += 256) {
        int t = e & 15, L = (e >> 4) & 31, nt = (e >> 9) & 7, kt = e >> 12;
        int i = kt * 32 + (L >> 4) * 16 + t;
        int o = nt * 16 + (L & 15);
        wp1[e] = f2bf(W1[o * 128 + i]);
        wp2[e] = f2bf(W2[o * 128 + i]);
    }
}

// ---------------------------------------------------------------------------
// K1: transpose (B,C,N) f32 -> (B,N,C) bf16 so per-point feature rows are
// 128B contiguous (gathers then hit whole cachelines; 8MB fits in L2).
// ---------------------------------------------------------------------------
__global__ void transpose_cn(const float* __restrict__ src, u16* __restrict__ dst) {
    const int total = BB * CC * NN;
    for (int e = blockIdx.x * blockDim.x + threadIdx.x; e < total; e += gridDim.x * blockDim.x) {
        int n = e & (NN - 1);
        int c = (e >> 12) & (CC - 1);
        int b = e >> 18;
        dst[((size_t)(b * NN + n)) * CC + c] = f2bf(src[e]);
    }
}

// ---------------------------------------------------------------------------
// K2: exact KNN (points2 of one batch resident in LDS, 48KB) + gather + build
// x0 (M x 160 bf16, zero padded). One thread per query, register top-16.
// ---------------------------------------------------------------------------
__global__ __launch_bounds__(256) void knn_build(const float* __restrict__ points1,
                                                 const float* __restrict__ points2,
                                                 const u16* __restrict__ f1t,
                                                 const u16* __restrict__ f2t,
                                                 u16* __restrict__ x0) {
    __shared__ float px[NN], py[NN], pz[NN];
    int b = blockIdx.x >> 4;
    int chunk = blockIdx.x & 15;
    const float* p2 = points2 + (size_t)b * 3 * NN;
    for (int j = threadIdx.x; j < NN; j += 256) {
        px[j] = p2[j]; py[j] = p2[NN + j]; pz[j] = p2[2 * NN + j];
    }
    __syncthreads();

    int n = chunk * 256 + threadIdx.x;
    const float* p1 = points1 + (size_t)b * 3 * NN;
    float qx = p1[n], qy = p1[NN + n], qz = p1[2 * NN + n];

    float bd[KK]; int bi[KK];
#pragma unroll
    for (int i = 0; i < KK; ++i) { bd[i] = 3.4e38f; bi[i] = 0; }
    float wmax = 3.4e38f; int wpos = 0;

    for (int j = 0; j < NN; ++j) {
        float dx = qx - px[j], dy = qy - py[j], dz = qz - pz[j];
        float d = dx * dx + dy * dy + dz * dz;
        if (d < wmax) {
#pragma unroll
            for (int i = 0; i < KK; ++i) {
                bool r = (i == wpos);
                bd[i] = r ? d : bd[i];
                bi[i] = r ? j : bi[i];
            }
            wmax = bd[0]; wpos = 0;
#pragma unroll
            for (int i = 1; i < KK; ++i)
                if (bd[i] > wmax) { wmax = bd[i]; wpos = i; }
        }
    }

    // cache f1 row (64 bf16 = 128B) once
    uint4 f1c[8];
    const uint4* f1p = (const uint4*)&f1t[((size_t)(b * NN + n)) * CC];
#pragma unroll
    for (int t = 0; t < 8; ++t) f1c[t] = f1p[t];

#pragma unroll
    for (int k = 0; k < KK; ++k)
        __builtin_prefetch(&f2t[((size_t)b * NN + bi[k]) * CC], 0, 1);

    size_t q = (size_t)(b * NN + n);
    uint4 z4; z4.x = 0; z4.y = 0; z4.z = 0; z4.w = 0;
    for (int k = 0; k < KK; ++k) {
        int j = bi[k];
        const uint4* f2p = (const uint4*)&f2t[((size_t)b * NN + j) * CC];
        uint4* d4 = (uint4*)&x0[(q * KK + k) * 160];
#pragma unroll
        for (int t = 0; t < 8; ++t) d4[t] = f2p[t];       // ch 0..63  feat_g
#pragma unroll
        for (int t = 0; t < 8; ++t) d4[8 + t] = f1c[t];   // ch 64..127 f1
        float rx = px[j] - qx, ry = py[j] - qy, rz = pz[j] - qz;
        uint4 tail;
        tail.x = (u32)f2bf(rx) | ((u32)f2bf(ry) << 16);
        tail.y = (u32)f2bf(rz);
        tail.z = 0; tail.w = 0;
        d4[16] = tail; d4[17] = z4; d4[18] = z4; d4[19] = z4;  // ch 128..159
    }
}

// ---------------------------------------------------------------------------
// K3/K5/K7: bf16 WMMA GEMM  H[M x 128] = act(A[M x KP]) * Wp^T + bias,
// with per-channel sum/sumsq accumulation for BatchNorm (LDS -> 2 global
// atomics per channel per block). AFFINE fuses prev layer's BN+ReLU into the
// A-tile LDS load. Block: 256 threads = 8 waves, tile 128(M) x 128(N), full K.
// ---------------------------------------------------------------------------
template<int KP, bool AFFINE>
__global__ __launch_bounds__(256) void gemm_bn(const u16* __restrict__ A,
                                               const u16* __restrict__ Wp,
                                               const float* __restrict__ bias,
                                               const float* __restrict__ ssPrev,
                                               u16* __restrict__ H,
                                               float* __restrict__ stat) {
    constexpr int KT = KP / 32;
    constexpr int SA = KP + 8;             // halfs; stride*2 is a multiple of 16B
    __shared__ __align__(16) u16 At[128 * SA];
    __shared__ __align__(16) u16 Wl[KT * 8 * 32 * 16];
    __shared__ float sSum[128], sSq[128];
    __shared__ float sc[AFFINE ? 128 : 1], sh[AFFINE ? 128 : 1];

    int tid = threadIdx.x;
    if (tid < 128) {
        sSum[tid] = 0.0f; sSq[tid] = 0.0f;
        if (AFFINE) { sc[tid] = ssPrev[tid]; sh[tid] = ssPrev[128 + tid]; }
    }
    __syncthreads();

    { // weights -> LDS (already in fragment order)
        const uint4* ws4 = (const uint4*)Wp;
        uint4* wl4 = (uint4*)Wl;
        constexpr int WC = KT * 8 * 32 * 16 / 8;
        for (int i = tid; i < WC; i += 256) wl4[i] = ws4[i];
    }
    { // A tile -> LDS, fusing BN affine + ReLU of previous layer
        constexpr int CH = KP / 8;
        size_t rowBase = (size_t)blockIdx.x * 128;
        for (int ck = tid; ck < 128 * CH; ck += 256) {
            int r = ck / CH;
            int cb = (ck % CH) * 8;
            uint4 v = *(const uint4*)&A[(rowBase + r) * KP + cb];
            if (AFFINE) {
                u32 w[4] = { v.x, v.y, v.z, v.w };
#pragma unroll
                for (int t = 0; t < 4; ++t) {
                    int c0 = cb + 2 * t, c1 = cb + 2 * t + 1;
                    float f0 = fmaxf(bf2f(w[t] & 0xFFFFu) * sc[c0] + sh[c0], 0.0f);
                    float f1 = fmaxf(bf2f(w[t] >> 16)     * sc[c1] + sh[c1], 0.0f);
                    w[t] = (u32)f2bf(f0) | ((u32)f2bf(f1) << 16);
                }
                v.x = w[0]; v.y = w[1]; v.z = w[2]; v.w = w[3];
            }
            *(uint4*)&At[r * SA + cb] = v;
        }
    }
    __syncthreads();

    int lane = tid & 31;
    int wave = tid >> 5;
    int m    = lane & 15;
    int off  = (lane >> 4) * 8;            // C/D row offset per half-wave; A-fragment K offset is 2*off
    int aRow = wave * 16 + m;

    v8f acc[8];
#pragma unroll
    for (int nt = 0; nt < 8; ++nt)
#pragma unroll
        for (int i = 0; i < 8; ++i) acc[nt][i] = 0.0f;

#pragma unroll
    for (int kt = 0; kt < KT; ++kt) {
        int kb = kt * 32 + 2 * off;
        Frag a;
        a.q[0] = *(const uint4*)&At[aRow * SA + kb];        // K {2*off..2*off+7}
        a.q[1] = *(const uint4*)&At[aRow * SA + kb + 8];    // K {2*off+8..2*off+15}
#pragma unroll
        for (int nt = 0; nt < 8; ++nt) {
            Frag bfr;
            const u16* wf = &Wl[(((kt * 8) + nt) * 32 + lane) * 16];
            bfr.q[0] = *(const uint4*)&wf[0];
            bfr.q[1] = *(const uint4*)&wf[8];
            acc[nt] = __builtin_amdgcn_wmma_f32_16x16x32_bf16(
                false, a.v, false, bfr.v, (short)0, acc[nt], false, false);
        }
    }

    // epilogue: add bias, emit bf16 H, per-channel stats
    size_t rowBase = (size_t)blockIdx.x * 128;
#pragma unroll
    for (int nt = 0; nt < 8; ++nt) {
        int ch = nt * 16 + m;
        float bv = bias[ch];
        float s = 0.0f, s2 = 0.0f;
#pragma unroll
        for (int i = 0; i < 8; ++i) {
            float v = acc[nt][i] + bv;
            s += v; s2 += v * v;
            size_t R = rowBase + (size_t)(wave * 16 + off + i);
            H[R * 128 + ch] = f2bf(v);
        }
        atomicAdd(&sSum[ch], s);
        atomicAdd(&sSq[ch], s2);
    }
    __syncthreads();
    if (tid < 128) {
        atomicAdd(&stat[tid], sSum[tid]);
        atomicAdd(&stat[128 + tid], sSq[tid]);
    }
}

// ---------------------------------------------------------------------------
// K4/K6/K8: fold BN stats into per-channel scale/shift.
// ---------------------------------------------------------------------------
__global__ void bn_finalize(const float* __restrict__ stat, const float* __restrict__ gamma,
                            const float* __restrict__ beta, float* __restrict__ ss) {
    int c = threadIdx.x;
    if (c < 128) {
        const float invM = 1.0f / (float)MTOT;
        float mu  = stat[c] * invM;
        float var = fmaxf(stat[128 + c] * invM - mu * mu, 0.0f);
        float scale = gamma[c] * rsqrtf(var + 1.0e-3f);
        ss[c] = scale;
        ss[128 + c] = beta[c] - mu * scale;
    }
}

// ---------------------------------------------------------------------------
// K9: apply last BN+ReLU, max over K=16, write (B,128,N) f32.
// ---------------------------------------------------------------------------
__global__ __launch_bounds__(128) void maxk_out(const u16* __restrict__ h2,
                                                const float* __restrict__ ss,
                                                float* __restrict__ out) {
    int q = blockIdx.x;            // b*N + n
    int c = threadIdx.x;
    float s = ss[c], t = ss[128 + c];
    const u16* p = h2 + (size_t)q * KK * 128 + c;
    float m = 0.0f;                // ReLU output >= 0
#pragma unroll
    for (int k = 0; k < KK; ++k) {
        float v = bf2f(p[k * 128]);
        v = fmaxf(v * s + t, 0.0f);
        m = fmaxf(m, v);
    }
    int b = q >> 12, n = q & (NN - 1);
    out[((size_t)b * 128 + c) * NN + n] = m;
}

// ---------------------------------------------------------------------------
// Workspace layout (bytes, 256-aligned)
// ---------------------------------------------------------------------------
#define WP0_OFF  0u
#define WP1_OFF  40960u
#define WP2_OFF  73728u
#define STAT_OFF 106496u           // 3 x (sum[128], sumsq[128]) f32
#define SS_OFF   109568u           // 3 x (scale[128], shift[128]) f32
#define F1T_OFF  112640u           // B*N*C bf16 = 4MB
#define F2T_OFF  4306944u
#define X0_OFF   8501248u          // M*160 bf16 = 160MB (reused as layer-2 out)
#define HA_OFF   176273408u        // M*128 bf16
#define HB_OFF   310491136u        // M*128 bf16

extern "C" void kernel_launch(void* const* d_in, const int* in_sizes, int n_in,
                              void* d_out, int out_size, void* d_ws, size_t ws_size,
                              hipStream_t stream) {
    (void)in_sizes; (void)n_in; (void)out_size; (void)ws_size;
    const float* points1   = (const float*)d_in[0];
    const float* points2   = (const float*)d_in[1];
    const float* features1 = (const float*)d_in[2];
    const float* features2 = (const float*)d_in[3];
    const float* W0 = (const float*)d_in[4];
    const float* b0 = (const float*)d_in[5];
    const float* g0 = (const float*)d_in[6];
    const float* be0 = (const float*)d_in[7];
    const float* W1 = (const float*)d_in[8];
    const float* b1 = (const float*)d_in[9];
    const float* g1 = (const float*)d_in[10];
    const float* be1 = (const float*)d_in[11];
    const float* W2 = (const float*)d_in[12];
    const float* b2 = (const float*)d_in[13];
    const float* g2 = (const float*)d_in[14];
    const float* be2 = (const float*)d_in[15];

    char* ws = (char*)d_ws;
    u16*   wp0  = (u16*)(ws + WP0_OFF);
    u16*   wp1  = (u16*)(ws + WP1_OFF);
    u16*   wp2  = (u16*)(ws + WP2_OFF);
    float* stat = (float*)(ws + STAT_OFF);
    float* ss   = (float*)(ws + SS_OFF);
    u16*   f1t  = (u16*)(ws + F1T_OFF);
    u16*   f2t  = (u16*)(ws + F2T_OFF);
    u16*   x0   = (u16*)(ws + X0_OFF);
    u16*   hA   = (u16*)(ws + HA_OFF);
    u16*   hB   = (u16*)(ws + HB_OFF);
    u16*   hC   = x0;   // layer-2 output reuses x0 region (stride 128 <= 160)

    prep_kernel<<<1, 256, 0, stream>>>(W0, W1, W2, wp0, wp1, wp2, stat);
    transpose_cn<<<8192, 256, 0, stream>>>(features1, f1t);
    transpose_cn<<<8192, 256, 0, stream>>>(features2, f2t);
    knn_build<<<BB * 16, 256, 0, stream>>>(points1, points2, f1t, f2t, x0);

    gemm_bn<160, false><<<MTOT / 128, 256, 0, stream>>>(x0, wp0, b0, nullptr, hA, stat);
    bn_finalize<<<1, 128, 0, stream>>>(stat, g0, be0, ss);

    gemm_bn<128, true><<<MTOT / 128, 256, 0, stream>>>(hA, wp1, b1, ss, hB, stat + 256);
    bn_finalize<<<1, 128, 0, stream>>>(stat + 256, g1, be1, ss + 256);

    gemm_bn<128, true><<<MTOT / 128, 256, 0, stream>>>(hB, wp2, b2, ss + 256, hC, stat + 512);
    bn_finalize<<<1, 128, 0, stream>>>(stat + 512, g2, be2, ss + 512);

    maxk_out<<<BB * NN, 128, 0, stream>>>(hC, ss + 512, (float*)d_out);
}
